// LatentHead_27410481283081
// MI455X (gfx1250) — matrix operand order repaired and checked by reading, not compiled, not run on hardware
//
#include <hip/hip_runtime.h>
#include <math.h>
#include <stdint.h>

typedef __attribute__((ext_vector_type(16))) _Float16 v16h;
typedef __attribute__((ext_vector_type(8)))  _Float16 v8h;
typedef __attribute__((ext_vector_type(8)))  float    v8f;
typedef __attribute__((ext_vector_type(4)))  unsigned int u32x4;
typedef __attribute__((ext_vector_type(8)))  int      i32x8;
typedef __attribute__((ext_vector_type(4)))  int      i32x4;

#define WMMA_F16(a, b, c) \
  __builtin_amdgcn_wmma_f32_16x16x32_f16(false, (a), false, (b), (short)0, (c), false, false)

#if __has_builtin(__builtin_amdgcn_tensor_load_to_lds) && \
    __has_builtin(__builtin_amdgcn_s_wait_tensorcnt)
#define USE_TDM 1
#else
#warning "CDNA5 TDM builtins unavailable -- flash_attn using cooperative-copy fallback"
#endif

static constexpr int Bz = 4;
static constexpr int T  = 4096;
static constexpr int D  = 1024;
static constexpr int H  = 64;

// ---------------------------------------------------------------------------
// Cross-lane reductions over the 16-lane N group (lane bits 0..3).
// ---------------------------------------------------------------------------
__device__ inline float rowmax16(float v) {
#pragma unroll
  for (int off = 8; off > 0; off >>= 1) v = fmaxf(v, __shfl_xor(v, off, 32));
  return v;
}
__device__ inline float rowsum16(float v) {
#pragma unroll
  for (int off = 8; off > 0; off >>= 1) v += __shfl_xor(v, off, 32);
  return v;
}

#ifdef USE_TDM
__device__ inline unsigned rfl(unsigned v) {
  return (unsigned)__builtin_amdgcn_readfirstlane((int)v);
}
// Issue one 2D TDM tile load: global (row-major, 2-byte elems) -> LDS.
// tile_d0 = contiguous elems per row, tile_d1 = rows, stride0 = row stride.
__device__ inline void tdm_load_2d(unsigned lds_addr, const void* gptr,
                                   unsigned tile_d0, unsigned tile_d1,
                                   unsigned long long stride0)
{
  const unsigned long long ga = (unsigned long long)(uintptr_t)gptr;
  u32x4 g0;
  g0[0] = rfl(1u);                                   // count=1, user mode
  g0[1] = rfl(lds_addr);                             // lds_addr
  g0[2] = rfl((unsigned)(ga & 0xffffffffu));         // global_addr[31:0]
  g0[3] = rfl(((unsigned)(ga >> 32) & 0x01ffffffu)   // global_addr[56:32]
              | 0x80000000u);                        // type=2 (bits 127:126)
  i32x8 g1;
  g1[0] = (int)rfl(0x00010000u);                     // data_size=1 (2 bytes)
  g1[1] = (int)rfl((tile_d0 & 0xffffu) << 16);       // tensor_dim0[15:0]
  g1[2] = (int)rfl((tile_d1 & 0xffffu) << 16);       // dim0 hi=0 | tensor_dim1
  g1[3] = (int)rfl((tile_d0 & 0xffffu) << 16);       // dim1 hi=0 | tile_dim0
  g1[4] = (int)rfl(tile_d1 & 0xffffu);               // tile_dim1 | tile_dim2=0
  g1[5] = (int)rfl((unsigned)(stride0 & 0xffffffffu));     // dim0_stride lo
  g1[6] = (int)rfl((unsigned)((stride0 >> 32) & 0xffffu)); // stride hi | s1 lo
  g1[7] = 0;
  const i32x4 z4 = {0, 0, 0, 0};
#if __clang_major__ >= 23
  const i32x8 z8 = {};
  __builtin_amdgcn_tensor_load_to_lds(g0, g1, z4, z4, z8, 0);
#else
  __builtin_amdgcn_tensor_load_to_lds(g0, g1, z4, z4, 0);
#endif
}
#endif

// ---------------------------------------------------------------------------
// Flash attention (defined first so the disasm snippet shows the TDM path).
// 8 waves/block (256 thr); wave owns 16 query rows, block covers 128, so each
// K/V tile is shared by 8 waves.  Keys streamed in 32-key blocks, TDM
// double-buffered into LDS.  Grid = B * T/128 = 128 blocks.
// ---------------------------------------------------------------------------
__global__ __launch_bounds__(256) void flash_attn(
    const _Float16* __restrict__ qf,
    const _Float16* __restrict__ kf,
    const _Float16* __restrict__ vt,   // [b][dim][t] transposed
    float* __restrict__ out)
{
  __shared__ __align__(32) _Float16 sK [2][32 * H];   // [key][dim]  4 KB x2
  __shared__ __align__(32) _Float16 sVt[2][H * 32];   // [dim][key]  4 KB x2
  __shared__ __align__(32) _Float16 sP [8][16 * 32];  // per-wave P  8 KB

  const int lane = threadIdx.x & 31;
  const int wid  = threadIdx.x >> 5;
  const int g    = lane >> 4;
  const int rl   = lane & 15;

  const int b    = blockIdx.x / (T / 128);
  const int q0b  = (blockIdx.x % (T / 128)) * 128;   // block's first query row
  const int q0   = q0b + wid * 16;                   // wave's first query row
  const int qmax = q0 + 15;

  // Q fragments (16x64 = two 16x32 A frags), vector global loads.
  const _Float16* qrow = qf + ((size_t)b * T + q0 + rl) * H;
  v16h Qfr[2];
#pragma unroll
  for (int c = 0; c < 2; ++c) {
    const v8h lo = *(const v8h*)(qrow + 32 * c + 16 * g);
    const v8h hi = *(const v8h*)(qrow + 32 * c + 16 * g + 8);
    Qfr[c] = __builtin_shufflevector(lo, hi, 0, 1, 2, 3, 4, 5, 6, 7,
                                             8, 9, 10, 11, 12, 13, 14, 15);
  }

  v8f acc[4] = {};
  float m[8], l[8];
#pragma unroll
  for (int r = 0; r < 8; ++r) { m[r] = -1e30f; l[r] = 0.0f; }

  const int nkb = q0b / 32 + 4;   // key blocks needed by the whole block

#ifdef USE_TDM
  if (wid == 0) {  // prologue: issue block 0 into buffer 0
    tdm_load_2d((unsigned)(uintptr_t)&sK[0][0],
                kf + ((size_t)b * T) * H, 64, 32, 64);
    tdm_load_2d((unsigned)(uintptr_t)&sVt[0][0],
                vt + (size_t)b * H * T, 32, 64, T);
  }
#endif

  for (int kbI = 0; kbI < nkb; ++kbI) {
    const int kbase = kbI * 32;
    const int cur   = kbI & 1;

#ifdef USE_TDM
    if (wid == 0) __builtin_amdgcn_s_wait_tensorcnt(0);
    __syncthreads();
    if (wid == 0 && kbI + 1 < nkb) {   // prefetch next block into other buffer
      tdm_load_2d((unsigned)(uintptr_t)&sK[cur ^ 1][0],
                  kf + ((size_t)b * T + kbase + 32) * H, 64, 32, 64);
      tdm_load_2d((unsigned)(uintptr_t)&sVt[cur ^ 1][0],
                  vt + (size_t)b * H * T + kbase + 32, 32, 64, T);
    }
#else
    __syncthreads();
    {
      const uint32_t* gk = (const uint32_t*)(kf + ((size_t)b * T + kbase) * H);
      const uint32_t* gv = (const uint32_t*)(vt + (size_t)b * H * T + kbase);
      uint32_t* dk = (uint32_t*)&sK[cur][0];
      uint32_t* dv = (uint32_t*)&sVt[cur][0];
#pragma unroll
      for (int j = 0; j < 4; ++j) {
        const int idx = (int)threadIdx.x + j * 256;
        dk[idx] = gk[idx];
        dv[idx] = gv[(idx >> 4) * (T / 2) + (idx & 15)];
      }
    }
    __syncthreads();
#endif

    if (kbase <= qmax) {
      const _Float16* sKc = &sK[cur][0];
      const _Float16* sVc = &sVt[cur][0];

      // ---- S = Q * K^T : contiguous 32B LDS fragment loads ----
      v8f S[2] = {};
#pragma unroll
      for (int t = 0; t < 2; ++t) {
        const int keyl = 16 * t + rl;
#pragma unroll
        for (int c = 0; c < 2; ++c) {
          const v16h bk = *(const v16h*)(sKc + keyl * H + 32 * c + 16 * g);
          S[t] = WMMA_F16(Qfr[c], bk, S[t]);
        }
      }

      // ---- scale + causal mask ----
#pragma unroll
      for (int t = 0; t < 2; ++t)
#pragma unroll
        for (int r = 0; r < 8; ++r) {
          const int qrowi = q0 + r + 8 * g;
          const int krowi = kbase + 16 * t + rl;
          const float s = S[t][r] * 0.125f;        // 1/sqrt(64)
          S[t][r] = (krowi <= qrowi) ? s : -1e30f;
        }

      // ---- online softmax ----
      float corr[8];
#pragma unroll
      for (int r = 0; r < 8; ++r) {
        const float mb = rowmax16(fmaxf(S[0][r], S[1][r]));
        const float mn = fmaxf(m[r], mb);
        corr[r] = __expf(m[r] - mn);
        m[r]    = mn;
      }
#pragma unroll
      for (int t = 0; t < 2; ++t)
#pragma unroll
        for (int r = 0; r < 8; ++r)
          S[t][r] = __expf(S[t][r] - m[r]);
#pragma unroll
      for (int r = 0; r < 8; ++r)
        l[r] = l[r] * corr[r] + rowsum16(S[0][r] + S[1][r]);
#pragma unroll
      for (int t = 0; t < 4; ++t)
#pragma unroll
        for (int r = 0; r < 8; ++r)
          acc[t][r] *= corr[r];

      // ---- P (C layout) -> LDS -> A layout ----
#pragma unroll
      for (int t = 0; t < 2; ++t)
#pragma unroll
        for (int r = 0; r < 8; ++r)
          sP[wid][(r + 8 * g) * 32 + 16 * t + rl] = (_Float16)S[t][r];

      asm volatile("s_wait_dscnt 0" ::: "memory");   // in-wave LDS RAW

      const v8h plo = *(const v8h*)(&sP[wid][0] + rl * 32 + 16 * g);
      const v8h phi = *(const v8h*)(&sP[wid][0] + rl * 32 + 16 * g + 8);
      const v16h pa = __builtin_shufflevector(plo, phi, 0, 1, 2, 3, 4, 5, 6, 7,
                                              8, 9, 10, 11, 12, 13, 14, 15);

      // ---- acc += P * V : contiguous 32B LDS fragment loads ----
#pragma unroll
      for (int t = 0; t < 4; ++t) {
        const v16h bv = *(const v16h*)(sVc + (16 * t + rl) * 32 + 16 * g);
        acc[t] = WMMA_F16(pa, bv, acc[t]);
      }
    }
  }

  // ---- epilogue: out = acc / l ----
#pragma unroll
  for (int t = 0; t < 4; ++t)
#pragma unroll
    for (int r = 0; r < 8; ++r) {
      const int qrowi = q0 + r + 8 * g;
      out[((size_t)b * T + qrowi) * H + 16 * t + rl] = acc[t][r] / l[r];
    }
}

// ---------------------------------------------------------------------------
// One-time weight prep: Wq/Wk/Wv (f32 [1024,64]) -> f16 B-fragment layout
// [kc(32)][w(3)][t(4)][lane(32)][e(16)].
// Fragment map: K = kc*32 + 16*(lane/16) + e,  N = 16*t + lane%16.
// ---------------------------------------------------------------------------
__global__ __launch_bounds__(32) void prep_w(
    const float* __restrict__ Wq, const float* __restrict__ Wk,
    const float* __restrict__ Wv, _Float16* __restrict__ wp)
{
  const int id   = blockIdx.x;            // 0..383 = kc*12 + w*4 + t
  const int kc   = id / 12;
  const int w    = (id % 12) / 4;
  const int t    = id % 4;
  const int lane = threadIdx.x;
  const int g    = lane >> 4;
  const int rl   = lane & 15;

  const float* W = (w == 0) ? Wq : (w == 1) ? Wk : Wv;

  v16h frag;
#pragma unroll
  for (int e = 0; e < 16; ++e)
    frag[e] = (_Float16)W[(size_t)(kc * 32 + 16 * g + e) * H + 16 * t + rl];

  *(v16h*)(wp + ((size_t)id * 32 + lane) * 16) = frag;
}

// ---------------------------------------------------------------------------
// q/k/v projections.  M = B*T = 16384, K = 1024, N = 64.
// One wave per 16-row tile, 4 waves/block -> 256 blocks of 128 threads.
// Q,K stored row-major f16; V stored transposed [b][dim][t] for flash PV.
// ---------------------------------------------------------------------------
__global__ __launch_bounds__(128) void proj_qkv(
    const float* __restrict__ x,
    const _Float16* __restrict__ wp,
    _Float16* __restrict__ qf,
    _Float16* __restrict__ kf,
    _Float16* __restrict__ vt)
{
  const int lane = threadIdx.x & 31;
  const int wid  = threadIdx.x >> 5;
  const int g    = lane >> 4;
  const int rl   = lane & 15;

  const int tile = blockIdx.x * 4 + wid;   // 0..1023
  const int row0 = tile * 16;

  v8f acc[3][4] = {};

  const float* xrow = x + (size_t)(row0 + rl) * D;

  for (int kc = 0; kc < 32; ++kc) {
    const int kb = kc * 32;

    // A fragment from x: 4 x float4 contiguous loads, convert to f16.
    const float4 a0 = *(const float4*)(xrow + kb + 16 * g);
    const float4 a1 = *(const float4*)(xrow + kb + 16 * g + 4);
    const float4 a2 = *(const float4*)(xrow + kb + 16 * g + 8);
    const float4 a3 = *(const float4*)(xrow + kb + 16 * g + 12);
    v16h a;
    a[0]  = (_Float16)a0.x; a[1]  = (_Float16)a0.y;
    a[2]  = (_Float16)a0.z; a[3]  = (_Float16)a0.w;
    a[4]  = (_Float16)a1.x; a[5]  = (_Float16)a1.y;
    a[6]  = (_Float16)a1.z; a[7]  = (_Float16)a1.w;
    a[8]  = (_Float16)a2.x; a[9]  = (_Float16)a2.y;
    a[10] = (_Float16)a2.z; a[11] = (_Float16)a2.w;
    a[12] = (_Float16)a3.x; a[13] = (_Float16)a3.y;
    a[14] = (_Float16)a3.z; a[15] = (_Float16)a3.w;

    const _Float16* wbase = wp + ((size_t)kc * 12 * 32 + lane) * 16;
#pragma unroll
    for (int w = 0; w < 3; ++w)
#pragma unroll
      for (int t = 0; t < 4; ++t) {
        const v16h bfr = *(const v16h*)(wbase + (size_t)(w * 4 + t) * 32 * 16);
        acc[w][t] = WMMA_F16(a, bfr, acc[w][t]);
      }
  }

  // C layout: VGPR r -> row (r + 8g), N = 16t + lane%16.
#pragma unroll
  for (int t = 0; t < 4; ++t)
#pragma unroll
    for (int r = 0; r < 8; ++r) {
      const int row = row0 + r + 8 * g;
      const int bb  = row >> 12;           // row / T
      const int ti  = row & (T - 1);
      qf[(size_t)row * H + 16 * t + rl] = (_Float16)acc[0][t][r];
      kf[(size_t)row * H + 16 * t + rl] = (_Float16)acc[1][t][r];
      vt[((size_t)bb * H + 16 * t + rl) * T + ti] = (_Float16)acc[2][t][r];
    }
}

// ---------------------------------------------------------------------------
extern "C" void kernel_launch(void* const* d_in, const int* in_sizes, int n_in,
                              void* d_out, int out_size, void* d_ws, size_t ws_size,
                              hipStream_t stream) {
  (void)in_sizes; (void)n_in; (void)out_size; (void)ws_size;
  const float* x  = (const float*)d_in[0];
  const float* Wq = (const float*)d_in[1];
  const float* Wk = (const float*)d_in[2];
  const float* Wv = (const float*)d_in[3];

  const size_t rows = (size_t)Bz * T * H;     // 1,048,576 elems
  _Float16* ws = (_Float16*)d_ws;
  _Float16* qf = ws;                          // 2 MB
  _Float16* kf = ws + rows;                   // 2 MB
  _Float16* vt = ws + 2 * rows;               // 2 MB (transposed V)
  _Float16* wp = ws + 3 * rows;               // 384 KB fragment-ready weights

  float* out = (float*)d_out;

  prep_w    <<<384, 32, 0, stream>>>(Wq, Wk, Wv, wp);
  proj_qkv  <<<(Bz * T) / 64, 128, 0, stream>>>(x, wp, qf, kf, vt);
  flash_attn<<<Bz * (T / 128), 256, 0, stream>>>(qf, kf, vt, out);
}
